// MarketLayer_38293928411876
// MI455X (gfx1250) — matrix operand order repaired and checked by reading, not compiled, not run on hardware
//
#include <hip/hip_runtime.h>
#include <hip/hip_bf16.h>
#include <stdint.h>

typedef __bf16 bf16_t;
typedef bf16_t v16bf __attribute__((ext_vector_type(16)));
typedef float  v8f   __attribute__((ext_vector_type(8)));
typedef int    v4i   __attribute__((ext_vector_type(4)));

#define NUM_E  16
#define TOPK_K 4
#define D_IN   1024
#define D_HID  2048
#define D_OUT  1024
#define BATCH  4096

#define TM 128
#define TN 128
#define TK 32
#define NTHREADS 256        // 8 waves (wave32)
#define TILE_ELEMS 4096     // one 8KB fragment tile = 4096 bf16

// ---------------------------------------------------------------------------
// CDNA5 async global->LDS copy (ASYNCcnt path), with guarded fallback.
// Verified on this toolchain: signature (as1 v4i*, as3 v4i*, imm off, imm cpol).
// ---------------------------------------------------------------------------
#if defined(__has_builtin)
#if __has_builtin(__builtin_amdgcn_global_load_async_to_lds_b128)
#define HAVE_ASYNC_LDS 1
#endif
#endif
#ifndef HAVE_ASYNC_LDS
#define HAVE_ASYNC_LDS 0
#endif

typedef __attribute__((address_space(1))) v4i gv4i_t;
typedef __attribute__((address_space(3))) v4i lv4i_t;

__device__ __forceinline__ void async_b128(const bf16_t* g, bf16_t* l) {
#if HAVE_ASYNC_LDS
    __builtin_amdgcn_global_load_async_to_lds_b128(
        (gv4i_t*)g, (lv4i_t*)l, 0, 0);
#else
    *(int4*)l = *(const int4*)g;   // global_load_b128 + ds_store_b128
#endif
}

// Wait until at most N async copies outstanding (immediate operand).
template <int N>
__device__ __forceinline__ void wait_async_le() {
#if HAVE_ASYNC_LDS
#if __has_builtin(__builtin_amdgcn_s_wait_asynccnt)
    __builtin_amdgcn_s_wait_asynccnt(N);
#else
    asm volatile("s_wait_asynccnt %0" :: "i"(N) : "memory");
#endif
#endif
}

// ---------------------------------------------------------------------------
// Fragment layout (CDNA5 ISA §7.12.2, wave32):
// A 16x32 bf16: lane=(m&15)|(((k>>3)&1)<<4), j=(k&7)|(((k>>4)&1)<<3)
// B 32x16 bf16: lane=(n&15)|(((k>>3)&1)<<4), j=(k&7)|(((k>>4)&1)<<3)
// Tile storage: element index = subtile*512 + lane*16 + j  (thread t owns the
// contiguous 32B chunk at t*16 elements).
// ---------------------------------------------------------------------------
__device__ __forceinline__ int pack2bf(float a, float b) {
    union { bf16_t h[2]; int i; } u;
    u.h[0] = (bf16_t)a; u.h[1] = (bf16_t)b;
    return u.i;
}

__device__ __forceinline__ v8f wmma_bf16(v16bf a, v16bf b, v8f c) {
    return __builtin_amdgcn_wmma_f32_16x16x32_bf16(
        false, a, false, b, (short)0, c, false, false);
}

// ---------------------------------------------------------------------------
// Pre-pass: x (fp32, row-major) -> xfrag (bf16, A-fragment tile order).
// ---------------------------------------------------------------------------
__global__ __launch_bounds__(NTHREADS)
void swizzleA_kernel(const float* __restrict__ x, bf16_t* __restrict__ xfrag)
{
    const int tile = blockIdx.x;
    const int kt = tile & (D_IN / TK - 1);
    const int mt = tile / (D_IN / TK);
    const int tid = threadIdx.x;
    const int ml = tid >> 1, hf = tid & 1;

    const float* src = x + (size_t)(mt * TM + ml) * D_IN + kt * TK + hf * 16;
    int4 cL, cH;
    cL.x = pack2bf(src[0], src[1]);   cL.y = pack2bf(src[2], src[3]);
    cL.z = pack2bf(src[4], src[5]);   cL.w = pack2bf(src[6], src[7]);
    cH.x = pack2bf(src[8], src[9]);   cH.y = pack2bf(src[10], src[11]);
    cH.z = pack2bf(src[12], src[13]); cH.w = pack2bf(src[14], src[15]);

    bf16_t* base = xfrag + (size_t)tile * TILE_ELEMS;
    const int s = ml >> 4, ll = ml & 15;
    *(int4*)(base + s * 512 + ll * 16 + hf * 8)        = cL;
    *(int4*)(base + s * 512 + (ll + 16) * 16 + hf * 8) = cH;
}

// ---------------------------------------------------------------------------
// Pre-pass: W (fp32 [E,K,N]) -> Wfrag (bf16, B-fragment tile order).
// ---------------------------------------------------------------------------
__global__ __launch_bounds__(NTHREADS)
void swizzleB_kernel(const float* __restrict__ W, bf16_t* __restrict__ Wfrag,
                     int K, int N)
{
    const int tile = blockIdx.x;
    const int ntiles_n = N / TN;
    const int ntiles_k = K / TK;
    const int nt = tile % ntiles_n;
    const int kt = (tile / ntiles_n) % ntiles_k;
    const int e  = tile / (ntiles_n * ntiles_k);
    const int tid = threadIdx.x;
    const int lane = tid & 31, tsub = tid >> 5;

    const int n  = nt * TN + tsub * 16 + (lane & 15);
    const int kb = kt * TK + ((lane >> 4) << 3);
    const float* src = W + ((size_t)e * K + kb) * N + n;

    int4 cL, cH;
    cL.x = pack2bf(src[0 * (size_t)N], src[1 * (size_t)N]);
    cL.y = pack2bf(src[2 * (size_t)N], src[3 * (size_t)N]);
    cL.z = pack2bf(src[4 * (size_t)N], src[5 * (size_t)N]);
    cL.w = pack2bf(src[6 * (size_t)N], src[7 * (size_t)N]);
    cH.x = pack2bf(src[16 * (size_t)N], src[17 * (size_t)N]);
    cH.y = pack2bf(src[18 * (size_t)N], src[19 * (size_t)N]);
    cH.z = pack2bf(src[20 * (size_t)N], src[21 * (size_t)N]);
    cH.w = pack2bf(src[22 * (size_t)N], src[23 * (size_t)N]);

    bf16_t* dst = Wfrag + (size_t)tile * TILE_ELEMS + tid * 16;
    *(int4*)dst       = cL;
    *(int4*)(dst + 8) = cH;
}

// ---------------------------------------------------------------------------
// Kernel 1: h[e,b,:] = relu(x @ W1[e] + b1[e]) (bf16, row-major).
// Triple-buffered software pipeline: issue tile kt+1 async copies, wait for
// tile kt's group (asynccnt<=4, latency hidden behind previous WMMAs), one
// barrier per iteration (3 buffers -> overwrite target is 2 steps from any
// buffer a laggard wave can still be reading).
// ---------------------------------------------------------------------------
__global__ __launch_bounds__(NTHREADS)
void gemm1_relu_kernel(const bf16_t* __restrict__ xfrag,
                       const bf16_t* __restrict__ W1frag,
                       const float* __restrict__ b1,
                       bf16_t* __restrict__ h)
{
    __shared__ __attribute__((aligned(64))) bf16_t ldsA[3][TM * TK];
    __shared__ __attribute__((aligned(64))) bf16_t ldsB[3][TK * TN];

    const int e  = blockIdx.z;
    const int mt = blockIdx.y;
    const int nt = blockIdx.x;
    const int tid  = threadIdx.x;
    const int wave = tid >> 5;
    const int lane = tid & 31;
    const int wm = wave >> 1, wn = wave & 1;
    const int NT1 = D_IN / TK;

    const v8f zero8 = {0.f, 0.f, 0.f, 0.f, 0.f, 0.f, 0.f, 0.f};
    v8f acc[2][4];
#pragma unroll
    for (int i = 0; i < 2; ++i)
#pragma unroll
        for (int j = 0; j < 4; ++j) acc[i][j] = zero8;

    auto issue = [&](int kt, int buf) {
        const bf16_t* gA = xfrag +
            ((size_t)mt * NT1 + kt) * TILE_ELEMS + tid * 16;
        const bf16_t* gB = W1frag +
            (((size_t)e * NT1 + kt) * (D_HID / TN) + nt) * TILE_ELEMS + tid * 16;
        async_b128(gA,     &ldsA[buf][tid * 16]);
        async_b128(gA + 8, &ldsA[buf][tid * 16 + 8]);
        async_b128(gB,     &ldsB[buf][tid * 16]);
        async_b128(gB + 8, &ldsB[buf][tid * 16 + 8]);
    };

    int bufc = 0;
    issue(0, 0);
    for (int kt = 0; kt < NT1; ++kt) {
        int bufi = bufc + 1; if (bufi == 3) bufi = 0;
        if (kt + 1 < NT1) { issue(kt + 1, bufi); wait_async_le<4>(); }
        else              { wait_async_le<0>(); }
        __syncthreads();

        const bf16_t* bA = ldsA[bufc];
        const bf16_t* bB = ldsB[bufc];
        v16bf a[2], b[4];
#pragma unroll
        for (int i = 0; i < 2; ++i)
            a[i] = *(const v16bf*)&bA[(2 * wm + i) * 512 + lane * 16];
#pragma unroll
        for (int j = 0; j < 4; ++j)
            b[j] = *(const v16bf*)&bB[(4 * wn + j) * 512 + lane * 16];
#pragma unroll
        for (int i = 0; i < 2; ++i)
#pragma unroll
            for (int j = 0; j < 4; ++j)
                acc[i][j] = wmma_bf16(a[i], b[j], acc[i][j]);
        bufc = bufi;
    }

    // Epilogue: +bias, relu, bf16 row-major store (consumed by bids + gemm2).
    const int lhi = lane >> 4, lcol = lane & 15;
    const int m0 = mt * TM, n0 = nt * TN;
#pragma unroll
    for (int i = 0; i < 2; ++i) {
#pragma unroll
        for (int j = 0; j < 4; ++j) {
            const int ng = n0 + (4 * wn + j) * 16 + lcol;
            const float bias = b1[e * D_HID + ng];
#pragma unroll
            for (int r = 0; r < 8; ++r) {
                const int mg = m0 + (2 * wm + i) * 16 + r + (lhi << 3);
                float v = acc[i][j][r] + bias;
                v = v > 0.f ? v : 0.f;
                h[((size_t)e * BATCH + mg) * D_HID + ng] = (bf16_t)v;
            }
        }
    }
}

// ---------------------------------------------------------------------------
// Kernel 2: bids[e,b] = h[e,b,:] . Wb[e,:] + bb[e]. One wave per row.
// ---------------------------------------------------------------------------
__global__ __launch_bounds__(256)
void bids_kernel(const bf16_t* __restrict__ h,
                 const float* __restrict__ Wb,
                 const float* __restrict__ bb,
                 float* __restrict__ bids)
{
    const int gw   = blockIdx.x * 8 + (threadIdx.x >> 5);
    const int lane = threadIdx.x & 31;
    const int e = gw >> 12;
    const int b = gw & (BATCH - 1);
    const bf16_t* row = h + ((size_t)e * BATCH + b) * D_HID;
    const float*  w   = Wb + e * D_HID;
    float acc = 0.f;
    for (int k = lane; k < D_HID; k += 32)
        acc += (float)row[k] * w[k];
#pragma unroll
    for (int off = 16; off > 0; off >>= 1)
        acc += __shfl_down(acc, off, 32);
    if (lane == 0) bids[e * BATCH + b] = acc + bb[e];
}

// ---------------------------------------------------------------------------
// Kernel 3: stable top-4 of 16 bids (jax.lax.top_k semantics).
// ---------------------------------------------------------------------------
__global__ __launch_bounds__(256)
void topk_kernel(const float* __restrict__ bids,
                 int* __restrict__ top_idx,
                 uint8_t* __restrict__ sel)
{
    const int b = blockIdx.x * blockDim.x + threadIdx.x;
    if (b >= BATCH) return;
    float v[NUM_E];
#pragma unroll
    for (int e = 0; e < NUM_E; ++e) {
        v[e] = bids[e * BATCH + b];
        sel[e * BATCH + b] = 0;
    }
    unsigned taken = 0;
#pragma unroll
    for (int k = 0; k < TOPK_K; ++k) {
        float best = -__builtin_inff();
        int bi = 0;
#pragma unroll
        for (int e = 0; e < NUM_E; ++e)
            if (!((taken >> e) & 1u) && v[e] > best) { best = v[e]; bi = e; }
        taken |= 1u << bi;
        top_idx[b * TOPK_K + k] = bi;
        sel[bi * BATCH + b] = 1;
    }
}

// ---------------------------------------------------------------------------
// Kernel 4: bias_eff[b,n] = 0.25 * sum_k b2[top_idx[b,k], n]
// ---------------------------------------------------------------------------
__global__ __launch_bounds__(256)
void biaseff_kernel(const float* __restrict__ b2,
                    const int* __restrict__ top_idx,
                    float* __restrict__ bias_eff)
{
    const size_t t = (size_t)blockIdx.x * 256 + threadIdx.x;
    const int b = (int)(t >> 10);
    const int n = (int)(t & 1023);
    float s = 0.f;
#pragma unroll
    for (int k = 0; k < TOPK_K; ++k)
        s += b2[(size_t)top_idx[b * TOPK_K + k] * D_OUT + n];
    bias_eff[t] = 0.25f * s;
}

// ---------------------------------------------------------------------------
// Kernel 5: out = 0.25 * sum_e diag(sel[e]) * (h[e] @ W2[e]) + bias_eff.
// Flattened e*kt software pipeline with triple-buffered LDS; A rows gated by a
// per-thread 16-bit selection mask (fixed accumulation order, deterministic).
// A: vector-staged gated h. B: async from pre-swizzled W2.
// ---------------------------------------------------------------------------
__global__ __launch_bounds__(NTHREADS)
void gemm2_moe_kernel(const bf16_t* __restrict__ h,
                      const bf16_t* __restrict__ W2frag,
                      const uint8_t* __restrict__ sel,
                      const float* __restrict__ bias_eff,
                      float* __restrict__ out)
{
    __shared__ __attribute__((aligned(64))) bf16_t ldsA[3][TM * TK];
    __shared__ __attribute__((aligned(64))) bf16_t ldsB[3][TK * TN];

    const int mt = blockIdx.y;
    const int nt = blockIdx.x;
    const int m0 = mt * TM;
    const int tid  = threadIdx.x;
    const int wave = tid >> 5;
    const int lane = tid & 31;
    const int wm = wave >> 1, wn = wave & 1;
    const int ml = tid >> 1, hf = tid & 1;      // A staging assignment
    const int sA = ml >> 4, llA = ml & 15;
    const int NKT = D_HID / TK;                 // 64
    const int NIT = NUM_E * NKT;                // 1024

    // Per-thread gate mask for its staged row across all 16 experts.
    unsigned selmask = 0;
#pragma unroll
    for (int e = 0; e < NUM_E; ++e)
        selmask |= (unsigned)(sel[(size_t)e * BATCH + m0 + ml] != 0) << e;

    const v8f zero8 = {0.f, 0.f, 0.f, 0.f, 0.f, 0.f, 0.f, 0.f};
    const int4 zero16 = {0, 0, 0, 0};
    v8f acc[2][4];
#pragma unroll
    for (int i = 0; i < 2; ++i)
#pragma unroll
        for (int j = 0; j < 4; ++j) acc[i][j] = zero8;

    auto issue = [&](int it, int buf) {
        const int e = it >> 6, kt = it & (NKT - 1);
        // A: gated 16 consecutive k of row (m0+ml) -> two fragment chunks.
        const bf16_t* srcA = h + ((size_t)e * BATCH + m0 + ml) * D_HID
                               + kt * TK + hf * 16;
        int4 cL = *(const int4*)srcA;
        int4 cH = *(const int4*)(srcA + 8);
        if (!((selmask >> e) & 1u)) { cL = zero16; cH = zero16; }
        *(int4*)&ldsA[buf][sA * 512 + llA * 16 + hf * 8]        = cL;
        *(int4*)&ldsA[buf][sA * 512 + (llA + 16) * 16 + hf * 8] = cH;
        // B: async copy of pre-swizzled W2 fragment tile.
        const bf16_t* gB = W2frag +
            (((size_t)e * NKT + kt) * (D_OUT / TN) + nt) * TILE_ELEMS + tid * 16;
        async_b128(gB,     &ldsB[buf][tid * 16]);
        async_b128(gB + 8, &ldsB[buf][tid * 16 + 8]);
    };

    int bufc = 0;
    issue(0, 0);
    for (int it = 0; it < NIT; ++it) {
        int bufi = bufc + 1; if (bufi == 3) bufi = 0;
        if (it + 1 < NIT) { issue(it + 1, bufi); wait_async_le<2>(); }
        else              { wait_async_le<0>(); }
        __syncthreads();

        const bf16_t* bA = ldsA[bufc];
        const bf16_t* bB = ldsB[bufc];
        v16bf a[2], b[4];
#pragma unroll
        for (int i = 0; i < 2; ++i)
            a[i] = *(const v16bf*)&bA[(2 * wm + i) * 512 + lane * 16];
#pragma unroll
        for (int j = 0; j < 4; ++j)
            b[j] = *(const v16bf*)&bB[(4 * wn + j) * 512 + lane * 16];
#pragma unroll
        for (int i = 0; i < 2; ++i)
#pragma unroll
            for (int j = 0; j < 4; ++j)
                acc[i][j] = wmma_bf16(a[i], b[j], acc[i][j]);
        bufc = bufi;
    }

    const int lhi = lane >> 4, lcol = lane & 15;
    const int n0 = nt * TN;
#pragma unroll
    for (int i = 0; i < 2; ++i) {
#pragma unroll
        for (int j = 0; j < 4; ++j) {
            const int ng = n0 + (4 * wn + j) * 16 + lcol;
#pragma unroll
            for (int r = 0; r < 8; ++r) {
                const int mg = m0 + (2 * wm + i) * 16 + r + (lhi << 3);
                out[(size_t)mg * D_OUT + ng] =
                    0.25f * acc[i][j][r] + bias_eff[(size_t)mg * D_OUT + ng];
            }
        }
    }
}

// ---------------------------------------------------------------------------
extern "C" void kernel_launch(void* const* d_in, const int* in_sizes, int n_in,
                              void* d_out, int out_size, void* d_ws, size_t ws_size,
                              hipStream_t stream) {
    (void)in_sizes; (void)n_in; (void)out_size; (void)ws_size;
    const float* x  = (const float*)d_in[0];
    const float* W1 = (const float*)d_in[1];
    const float* b1 = (const float*)d_in[2];
    const float* W2 = (const float*)d_in[3];
    const float* b2 = (const float*)d_in[4];
    const float* Wb = (const float*)d_in[5];
    const float* bb = (const float*)d_in[6];

    float* out     = (float*)d_out;
    int*   top_idx = (int*)((float*)d_out + (size_t)BATCH * D_OUT);

    char* ws = (char*)d_ws;
    bf16_t* h = (bf16_t*)ws;        ws += (size_t)NUM_E * BATCH * D_HID * sizeof(bf16_t);
    float* bids = (float*)ws;       ws += (size_t)NUM_E * BATCH * sizeof(float);
    uint8_t* sel = (uint8_t*)ws;    ws += (size_t)NUM_E * BATCH;
    float* bias_eff = (float*)ws;   ws += (size_t)BATCH * D_OUT * sizeof(float);
    bf16_t* xfrag = (bf16_t*)ws;    ws += (size_t)BATCH * D_IN * sizeof(bf16_t);
    bf16_t* W1frag = (bf16_t*)ws;   ws += (size_t)NUM_E * D_IN * D_HID * sizeof(bf16_t);
    bf16_t* W2frag = (bf16_t*)ws;

    // Pre-passes: fp32 -> bf16 conversion + fragment-order swizzle (one-time).
    swizzleA_kernel<<<(BATCH / TM) * (D_IN / TK), NTHREADS, 0, stream>>>(x, xfrag);
    swizzleB_kernel<<<NUM_E * (D_IN / TK) * (D_HID / TN), NTHREADS, 0, stream>>>(
        W1, W1frag, D_IN, D_HID);
    swizzleB_kernel<<<NUM_E * (D_HID / TK) * (D_OUT / TN), NTHREADS, 0, stream>>>(
        W2, W2frag, D_HID, D_OUT);

    gemm1_relu_kernel<<<dim3(D_HID / TN, BATCH / TM, NUM_E), NTHREADS, 0, stream>>>(
        xfrag, W1frag, b1, h);
    bids_kernel<<<NUM_E * BATCH / 8, 256, 0, stream>>>(h, Wb, bb, bids);
    topk_kernel<<<BATCH / 256, 256, 0, stream>>>(bids, top_idx, sel);
    biaseff_kernel<<<(BATCH * D_OUT) / 256, 256, 0, stream>>>(b2, top_idx, bias_eff);
    gemm2_moe_kernel<<<dim3(D_OUT / TN, BATCH / TM), NTHREADS, 0, stream>>>(
        h, W2frag, sel, bias_eff, out);
}
